// LSAMixer_74826920231028
// MI455X (gfx1250) — matrix operand order repaired and checked by reading, not compiled, not run on hardware
//
#include <hip/hip_runtime.h>
#include <hip/hip_bf16.h>
#include <math.h>

// ---------------------------------------------------------------------------
// LSAMixer for MI455X (gfx1250, wave32, WMMA).
// All pointwise (1x1) convs are GEMMs D = W[ Cout x Cin ] * X[ Cin x N ] done
// with v_wmma_f32_16x16x32_f16 (f16 inputs, f32 accumulate).
// ---------------------------------------------------------------------------

#define NTOT 16384   // H*W
#define HH   128
#define WW   128

typedef __attribute__((ext_vector_type(16))) _Float16 v16h;
typedef __attribute__((ext_vector_type(8)))  float    v8f;

// ---------------------------------------------------------------------------
// Pointwise conv as WMMA GEMM, M-blocked: each wave computes MT*16 rows x 64
// cols; the 4 B fragments per K-step are loaded once and reused against MT A
// fragments (weights are tiny and L2/L0 hot), cutting activation (B) traffic
// MT-fold versus one-tile-per-wave.
//   out[b, o, n] = act( sum_c w[o*Cin+c] * in[b*inB + c*N + n] + bias[o] )
// grid = (N/512, ceil(Cout/(MT*16)), B), block = 256 (8 waves, 64 cols each).
// Fragment layouts used here (A/B: contiguous K order, same for both):
//   A (16x32 f16): lane m=lane&15, half=lane>>4; element e -> K=half*16+e
//   B (32x16 f16): lane n=lane&15, half=lane>>4; element e -> K=half*16+e
//   C/D (16x16 f32): element r -> row = half*8 + r, col = lane&15
// ---------------------------------------------------------------------------
template<int MT>
__global__ __launch_bounds__(256) void pw_gemm_wmma(
    const float* __restrict__ in, const float* __restrict__ wmat,
    const float* __restrict__ bias, float* __restrict__ out,
    int Cin, int Cout, long inB, long outB, int act)
{
    const int lane = threadIdx.x & 31;
    const int wave = threadIdx.x >> 5;
    const int m0   = blockIdx.y * (MT * 16);
    const int n0   = blockIdx.x * 512 + wave * 64;
    const int b    = blockIdx.z;
    const int m    = lane & 15;
    const int half = lane >> 4;

    const float* X = in + (size_t)b * inB;

    // Per-M-tile weight row pointers (clamped when OOB; zeroed after load).
    const float* wr[MT];
    bool rowOK[MT];
#pragma unroll
    for (int mt = 0; mt < MT; ++mt) {
        const int row = m0 + mt * 16 + m;
        rowOK[mt] = (row < Cout);
        wr[mt] = wmat + (size_t)(rowOK[mt] ? row : 0) * Cin;
    }

    v8f acc[MT][4] = {};

    for (int k0 = 0; k0 < Cin; k0 += 32) {
        // ---- 4 B fragments (activations), loaded once per K-step ----
        v16h bf[4];
#pragma unroll
        for (int s = 0; s < 4; ++s) {
            const float* pb = X + (size_t)(k0 + half * 16) * NTOT + (n0 + s * 16 + m);
#pragma unroll
            for (int e = 0; e < 16; ++e)
                bf[s][e] = (_Float16)pb[(size_t)e * NTOT];
        }
        // ---- MT A fragments, 16 WMMAs total per K-step (MT=4) ----
#pragma unroll
        for (int mt = 0; mt < MT; ++mt) {
            v16h a;
            const float* pa = wr[mt] + k0 + half * 16;
#pragma unroll
            for (int e = 0; e < 8; ++e) {
                float v0 = pa[e];
                float v1 = pa[e + 8];
                if (!rowOK[mt]) { v0 = 0.f; v1 = 0.f; }
                a[e]     = (_Float16)v0;
                a[e + 8] = (_Float16)v1;
            }
#pragma unroll
            for (int s = 0; s < 4; ++s)
                acc[mt][s] = __builtin_amdgcn_wmma_f32_16x16x32_f16(
                    false, a, false, bf[s], (short)0, acc[mt][s], false, false);
        }
    }

    // ---- epilogue: bias + activation + store ----
    float* O = out + (size_t)b * outB;
#pragma unroll
    for (int mt = 0; mt < MT; ++mt) {
#pragma unroll
        for (int s = 0; s < 4; ++s) {
            const int col = n0 + s * 16 + m;
#pragma unroll
            for (int r = 0; r < 8; ++r) {
                const int orow = m0 + mt * 16 + half * 8 + r;
                if (orow < Cout) {
                    float v = acc[mt][s][r] + bias[orow];
                    if (act == 1)      v = fmaxf(v, 0.f);
                    else if (act == 2) v = tanhf(v);
                    O[(size_t)orow * NTOT + col] = v;
                }
            }
        }
    }
}

// ---------------------------------------------------------------------------
// Depthwise conv, 1x7 (horizontal, vertical=0) or 7x1 (vertical=1), zero pad 3.
// in: [B,128,H,W] with batch stride inB; out: [B,128,H,W] contiguous.
// ---------------------------------------------------------------------------
__global__ __launch_bounds__(256) void dw_conv(
    const float* __restrict__ in, const float* __restrict__ w7,
    const float* __restrict__ bias, float* __restrict__ out,
    long inB, int vertical)
{
    const int t = blockIdx.x * 256 + threadIdx.x;
    const int x = t & (WW - 1);
    const int y = (t >> 7) & (HH - 1);
    const int c = (t >> 14) & 127;
    const int b = t >> 21;
    const float* p = in + (size_t)b * inB + (size_t)c * NTOT;
    float s = bias[c];
#pragma unroll
    for (int j = 0; j < 7; ++j) {
        const float wv = w7[c * 7 + j];
        if (vertical) {
            const int yy = y + j - 3;
            if (yy >= 0 && yy < HH) s += p[yy * WW + x] * wv;
        } else {
            const int xx = x + j - 3;
            if (xx >= 0 && xx < WW) s += p[y * WW + xx] * wv;
        }
    }
    out[(size_t)b * (128L * NTOT) + (size_t)c * NTOT + y * WW + x] = s;
}

// ---------------------------------------------------------------------------
// Fused sliding attention (_fsa). 5 taps from offsets [-3..3]:
//   t0=(x[-3]+x[-2])/2, t1=x[-1], t2=x[0], t3=x[+1], t4=(x[+2]+x[+3])/2
// out[b,c,.] = sum_k tap_k * kern[b, (c>>4)*5+k, .]
// ---------------------------------------------------------------------------
__global__ __launch_bounds__(256) void fsa_kernel(
    const float* __restrict__ x, const float* __restrict__ kern,
    float* __restrict__ out, long xB, int vertical)
{
    const int t = blockIdx.x * 256 + threadIdx.x;
    const int w = t & (WW - 1);
    const int h = (t >> 7) & (HH - 1);
    const int c = (t >> 14) & 127;
    const int b = t >> 21;
    const float* p = x + (size_t)b * xB + (size_t)c * NTOT;

    float sm[7];
#pragma unroll
    for (int j = 0; j < 7; ++j) {
        const int off = j - 3;
        float v = 0.f;
        if (vertical) {
            const int hh = h + off;
            if (hh >= 0 && hh < HH) v = p[hh * WW + w];
        } else {
            const int ww = w + off;
            if (ww >= 0 && ww < WW) v = p[h * WW + ww];
        }
        sm[j] = v;
    }
    float tap[5];
    tap[0] = 0.5f * (sm[0] + sm[1]);
    tap[1] = sm[2];
    tap[2] = sm[3];
    tap[3] = sm[4];
    tap[4] = 0.5f * (sm[5] + sm[6]);

    const int g = c >> 4;
    const float* kp = kern + (size_t)b * (40L * NTOT) + (size_t)(g * 5) * NTOT + h * WW + w;
    float s = 0.f;
#pragma unroll
    for (int k = 0; k < 5; ++k) s += tap[k] * kp[(size_t)k * NTOT];
    out[(size_t)b * (128L * NTOT) + (size_t)c * NTOT + h * WW + w] = s;
}

// ---------------------------------------------------------------------------
// at_mean: out[bc] = mean_n in[bc*N + n]     (grid = B*128 blocks)
// ---------------------------------------------------------------------------
__global__ __launch_bounds__(256) void row_mean(
    const float* __restrict__ in, float* __restrict__ out)
{
    __shared__ float red[256];
    const float* p = in + (size_t)blockIdx.x * NTOT;
    float s = 0.f;
    for (int i = threadIdx.x; i < NTOT; i += 256) s += p[i];
    red[threadIdx.x] = s;
    __syncthreads();
    for (int st = 128; st > 0; st >>= 1) {
        if (threadIdx.x < (unsigned)st) red[threadIdx.x] += red[threadIdx.x + st];
        __syncthreads();
    }
    if (threadIdx.x == 0) out[blockIdx.x] = red[0] * (1.f / NTOT);
}

// ---------------------------------------------------------------------------
// Fused: xc_mean[b,n] = mean_c xconv  ;  s_sig[b,n] = sigmoid(dot(at_mean,xc)/C2)
// One thread per (b,n). grid = B*N/256.
// ---------------------------------------------------------------------------
__global__ __launch_bounds__(256) void ssig_xcmean_kernel(
    const float* __restrict__ xconv, const float* __restrict__ atmean,
    float* __restrict__ xcmean, float* __restrict__ ssig)
{
    const int t = blockIdx.x * 256 + threadIdx.x;
    const int b = t >> 14;
    const int n = t & (NTOT - 1);
    const float* p  = xconv + (size_t)b * (128L * NTOT) + n;
    const float* am = atmean + b * 128;
    float sum = 0.f, dot = 0.f;
#pragma unroll 4
    for (int c = 0; c < 128; ++c) {
        const float v = p[(size_t)c * NTOT];
        sum += v;
        dot += v * am[c];
    }
    xcmean[t] = sum * (1.f / 128.f);
    ssig[t]   = 1.f / (1.f + expf(-dot * (1.f / 128.f)));
}

// ---------------------------------------------------------------------------
// c_sig[b,c] = sigmoid( sum_n attn[b,c,n]*xc_mean[b,n] / N )  (grid = B*128)
// ---------------------------------------------------------------------------
__global__ __launch_bounds__(256) void csig_kernel(
    const float* __restrict__ attn, const float* __restrict__ xcmean,
    float* __restrict__ csig)
{
    __shared__ float red[256];
    const int bc = blockIdx.x;
    const int b  = bc >> 7;
    const float* p = attn + (size_t)bc * NTOT;
    const float* q = xcmean + (size_t)b * NTOT;
    float s = 0.f;
    for (int i = threadIdx.x; i < NTOT; i += 256) s += p[i] * q[i];
    red[threadIdx.x] = s;
    __syncthreads();
    for (int st = 128; st > 0; st >>= 1) {
        if (threadIdx.x < (unsigned)st) red[threadIdx.x] += red[threadIdx.x + st];
        __syncthreads();
    }
    if (threadIdx.x == 0)
        csig[bc] = 1.f / (1.f + expf(-red[0] * (1.f / NTOT)));
}

// ---------------------------------------------------------------------------
// cat[b, c, n] = c<128 ? s_sig[b,n]*xconv[b,c,n] : c_sig[b,c-128]*attn[b,c-128,n]
// ---------------------------------------------------------------------------
__global__ __launch_bounds__(256) void gate_cat_kernel(
    const float* __restrict__ xconv, const float* __restrict__ attn,
    const float* __restrict__ ssig, const float* __restrict__ csig,
    float* __restrict__ cat)
{
    const long t = (long)blockIdx.x * 256 + threadIdx.x;   // b*256*N + c*N + n
    const int n = (int)(t & (NTOT - 1));
    const int c = (int)((t >> 14) & 255);
    const int b = (int)(t >> 22);
    float v;
    if (c < 128)
        v = ssig[b * NTOT + n] * xconv[(size_t)b * (128L * NTOT) + (size_t)c * NTOT + n];
    else
        v = csig[b * 128 + (c - 128)] *
            attn[(size_t)b * (128L * NTOT) + (size_t)(c - 128) * NTOT + n];
    cat[t] = v;
}

// ---------------------------------------------------------------------------
extern "C" void kernel_launch(void* const* d_in, const int* in_sizes, int n_in,
                              void* d_out, int out_size, void* d_ws, size_t ws_size,
                              hipStream_t stream)
{
    const float* x    = (const float*)d_in[0];
    const float* piw  = (const float*)d_in[1];
    const float* pib  = (const float*)d_in[2];
    const float* hkdw = (const float*)d_in[3];
    const float* hkdb = (const float*)d_in[4];
    const float* hkpw = (const float*)d_in[5];
    const float* hkpb = (const float*)d_in[6];
    const float* vkdw = (const float*)d_in[7];
    const float* vkdb = (const float*)d_in[8];
    const float* vkpw = (const float*)d_in[9];
    const float* vkpb = (const float*)d_in[10];
    const float* cbw1 = (const float*)d_in[11];
    const float* cbb1 = (const float*)d_in[12];
    const float* cbw2 = (const float*)d_in[13];
    const float* cbb2 = (const float*)d_in[14];
    const float* pow_ = (const float*)d_in[15];
    const float* pob  = (const float*)d_in[16];
    float* out = (float*)d_out;

    const long N = NTOT;
    char* ws = (char*)d_ws;
    // Workspace layout (aliased; all fp32):
    //   [  0,  64Mi)  h   (B x 256 x N), reused as `cat` for proj_out input
    //   [ 64,  96Mi)  xconv (B x 128 x N)
    //   [ 96, 128Mi)  attn_h (B x 128 x N)
    //   [128, 160Mi)  dwbuf (B x 128 x N), reused as `attn`
    //   [160, 176Mi)  t1 (B x 64 x N), reused as kh/kv (B x 40 x N)
    //   [176Mi, ...)  small stats
    float* h      = (float*)(ws);
    float* xconv  = (float*)(ws + ( 64L << 20));
    float* attnh  = (float*)(ws + ( 96L << 20));
    float* dwb    = (float*)(ws + (128L << 20));
    float* attn   = dwb;                         // alias: dwb dead after step 8
    float* t1     = (float*)(ws + (160L << 20));
    float* kb     = t1;                          // alias: t1 dead after step 3
    float* atmean = (float*)(ws + (176L << 20));
    float* xcmean = (float*)(ws + (176L << 20) + (1L << 20));
    float* ssig   = (float*)(ws + (176L << 20) + (2L << 20));
    float* csig   = (float*)(ws + (176L << 20) + (3L << 20));

    const dim3 blk(256);

    // 1) proj_in: 256 -> 256   (4 M-tiles/wave -> grid.y = 256/64 = 4)
    pw_gemm_wmma<4><<<dim3(32, 4, 4), blk, 0, stream>>>(x, piw, pib, h,
        256, 256, 256 * N, 256 * N, 0);
    // 2) conv-branch MLP: x1(128) -> 64, ReLU   (x1 = h, batch stride 256N)
    pw_gemm_wmma<4><<<dim3(32, 1, 4), blk, 0, stream>>>(h, cbw1, cbb1, t1,
        128, 64, 256 * N, 64 * N, 1);
    // 3) 64 -> 128  => xconv
    pw_gemm_wmma<4><<<dim3(32, 2, 4), blk, 0, stream>>>(t1, cbw2, cbb2, xconv,
        64, 128, 64 * N, 128 * N, 0);
    // 4) depthwise 1x7 along W on x2 (= h + 128N, batch stride 256N)
    dw_conv<<<32768, blk, 0, stream>>>(h + 128 * N, hkdw, hkdb, dwb, 256 * N, 0);
    // 5) kh = tanh(pw 128 -> 40)   (3 M-tiles cover 48 >= 40 rows)
    pw_gemm_wmma<3><<<dim3(32, 1, 4), blk, 0, stream>>>(dwb, hkpw, hkpb, kb,
        128, 40, 128 * N, 40 * N, 2);
    // 6) FSA along W on x2 => attn_h
    fsa_kernel<<<32768, blk, 0, stream>>>(h + 128 * N, kb, attnh, 256 * N, 0);
    // 7) depthwise 7x1 along H on attn_h
    dw_conv<<<32768, blk, 0, stream>>>(attnh, vkdw, vkdb, dwb, 128 * N, 1);
    // 8) kv = tanh(pw 128 -> 40)
    pw_gemm_wmma<3><<<dim3(32, 1, 4), blk, 0, stream>>>(dwb, vkpw, vkpb, kb,
        128, 40, 128 * N, 40 * N, 2);
    // 9) FSA along H on attn_h => attn (overwrites dwb region; dwb now dead)
    fsa_kernel<<<32768, blk, 0, stream>>>(attnh, kb, attn, 128 * N, 1);
    // 10) at_mean[b,c]
    row_mean<<<512, blk, 0, stream>>>(attn, atmean);
    // 11) xc_mean[b,n] + s_sig[b,n]
    ssig_xcmean_kernel<<<256, blk, 0, stream>>>(xconv, atmean, xcmean, ssig);
    // 12) c_sig[b,c]
    csig_kernel<<<512, blk, 0, stream>>>(attn, xcmean, csig);
    // 13) gated concat -> reuse h as proj_out input
    gate_cat_kernel<<<65536, blk, 0, stream>>>(xconv, attn, ssig, csig, h);
    // 14) proj_out: 256 -> 256 => d_out
    pw_gemm_wmma<4><<<dim3(32, 4, 4), blk, 0, stream>>>(h, pow_, pob, out,
        256, 256, 256 * N, 256 * N, 0);

    (void)in_sizes; (void)n_in; (void)out_size; (void)ws_size;
}
